// pure_EPI_dense_15891378995322
// MI455X (gfx1250) — matrix operand order repaired and checked by reading, not compiled, not run on hardware
//
#include <hip/hip_runtime.h>
#include <hip/hip_bf16.h>
#include <stdint.h>

typedef __attribute__((ext_vector_type(16))) _Float16 v16h;
typedef __attribute__((ext_vector_type(8)))  float    v8f;
typedef int epi_v4i __attribute__((vector_size(16)));   // matches builtin param type

#define N_NODES 4096
#define BATCH   8
#define WAVES   4
#define TILE_M  16
#define KSUPER  (WAVES * 32)     // 128 floats of K staged per superchunk
#define LDS_STRIDE 132           // 128 + 4 pad DWORDs -> conflict-free row access

// ---- gfx1250 async-to-LDS path (guarded so compile never breaks) ----------
#if defined(__has_builtin)
#  if __has_builtin(__builtin_amdgcn_global_load_async_to_lds_b128)
#    define EPI_ASYNC_LDS 1
#  endif
#  if __has_builtin(__builtin_amdgcn_s_wait_asynccnt)
#    define EPI_WAIT_BUILTIN 1
#  endif
#endif
#ifndef EPI_ASYNC_LDS
#  define EPI_ASYNC_LDS 0
#endif

__device__ __forceinline__ void epi_wait_async_le4() {
#if EPI_ASYNC_LDS
#  if defined(EPI_WAIT_BUILTIN)
    __builtin_amdgcn_s_wait_asynccnt(4);
#  else
    asm volatile("s_wait_asynccnt 0x4" ::: "memory");
#  endif
#endif
}
__device__ __forceinline__ void epi_wait_async_0() {
#if EPI_ASYNC_LDS
#  if defined(EPI_WAIT_BUILTIN)
    __builtin_amdgcn_s_wait_asynccnt(0);
#  else
    asm volatile("s_wait_asynccnt 0x0" ::: "memory");
#  endif
#endif
}

// Stage one 16 x 128 fp32 tile of W into LDS (4 b128 per thread, coalesced).
__device__ __forceinline__ void epi_stage_tile(const float* __restrict__ gsrc,
                                               float* __restrict__ lbuf) {
#pragma unroll
    for (int pass = 0; pass < 4; ++pass) {
        int q   = (int)threadIdx.x + pass * (WAVES * 32);
        int row = q >> 5;              // 0..15
        int col = (q & 31) << 2;       // 0..124, step 4
        const float* g = gsrc + (size_t)row * N_NODES + col;
        float*       l = lbuf + row * LDS_STRIDE + col;
#if EPI_ASYNC_LDS
        __builtin_amdgcn_global_load_async_to_lds_b128(
            (__attribute__((address_space(1))) epi_v4i*)(uintptr_t)g,
            (__attribute__((address_space(3))) epi_v4i*)(unsigned int)(uintptr_t)l,
            0, 0);
#else
        *(float4*)l = *(const float4*)g;
#endif
    }
}

// ---------------------------------------------------------------------------
// Workspace layout (bytes):
//   [0,        131072)  float    ps1[8][4096]
//   [131072,   262144)  _Float16 IP1[16][4096]   (I^1, batches 8..15 zeroed)
//   [262144,   393216)  _Float16 IP2[16][4096]   (I^2)
//   [393216,   524288)  _Float16 IP3[16][4096]   (I^3)
// ---------------------------------------------------------------------------

__global__ void epi_prep_powers(const float* __restrict__ state,
                                _Float16* __restrict__ IP1,
                                _Float16* __restrict__ IP2,
                                _Float16* __restrict__ IP3) {
    int idx = blockIdx.x * blockDim.x + threadIdx.x;      // over 16*4096
    if (idx >= 16 * N_NODES) return;
    int b = idx >> 12;
    int k = idx & (N_NODES - 1);
    float i1 = 0.0f;
    if (b < BATCH) i1 = state[(size_t)b * 4 * N_NODES + 2 * N_NODES + k];
    float i2 = i1 * i1;
    IP1[idx] = (_Float16)i1;
    IP2[idx] = (_Float16)i2;
    IP3[idx] = (_Float16)(i2 * i1);
}

// One block per 16-row tile of W. Double-buffered async staging of W into
// LDS; 4 waves each consume a 32-wide K slice of the staged 128-wide chunk.
// Y_p[j,b] = sum_k W^p[j,k] * I^p[b,k] accumulated in fp32 via WMMA.
__global__ __launch_bounds__(WAVES * 32)
void epi_infection_wmma(const float* __restrict__ W,
                        const float* __restrict__ state,
                        const float* __restrict__ psMatrix,
                        const _Float16* __restrict__ IP1,
                        const _Float16* __restrict__ IP2,
                        const _Float16* __restrict__ IP3,
                        float* __restrict__ ps1out) {
    __shared__ float ldsW[2][TILE_M * LDS_STRIDE];
    __shared__ float red[3][WAVES][8][32];

    const int lane   = threadIdx.x & 31;
    const int wave   = threadIdx.x >> 5;
    const int j0     = blockIdx.x * TILE_M;

    const int mrow   = lane & 15;             // A-matrix M row within tile
    const int hiHalf = (lane >> 4) & 1;       // 0: lanes 0-15, 1: lanes 16-31
    const int aBase  = hiHalf * 16;           // A layout: K 0-15 / 16-31
    const int bBase  = hiHalf * 16;           // B layout: K 0-15 / 16-31
    const int bcol   = lane & 15;             // B-matrix N column (= batch index)
    const int klocal = wave * 32;             // this wave's K slice in superchunk

    const float* wtile = W + (size_t)j0 * N_NODES;

    v8f acc1 = {}, acc2 = {}, acc3 = {};

    epi_stage_tile(wtile, ldsW[0]);
    int cur = 0;

    for (int sc = 0; sc < N_NODES; sc += KSUPER, cur ^= 1) {
        const bool more = (sc + KSUPER) < N_NODES;
        if (more) epi_stage_tile(wtile + sc + KSUPER, ldsW[cur ^ 1]);
        if (more) epi_wait_async_le4();
        else      epi_wait_async_0();
        __syncthreads();   // staged tile visible to all waves

        // ---- A operand from LDS (fp32 -> fp16, plus ^2, ^3) ----
        const float* lrow = &ldsW[cur][0] + mrow * LDS_STRIDE + klocal + aBase;
        const float4* lp0 = (const float4*)(lrow);
        const float4* lp1 = (const float4*)(lrow + 8);
        float4 f0 = lp0[0];
        float4 f1 = lp0[1];
        float4 f2 = lp1[0];
        float4 f3 = lp1[1];
        float wv[16] = { f0.x, f0.y, f0.z, f0.w,  f1.x, f1.y, f1.z, f1.w,
                         f2.x, f2.y, f2.z, f2.w,  f3.x, f3.y, f3.z, f3.w };
        v16h A1, A2, A3;
#pragma unroll
        for (int t = 0; t < 16; ++t) {
            float w1 = wv[t];
            float w2 = w1 * w1;
            A1[t] = (_Float16)w1;
            A2[t] = (_Float16)w2;
            A3[t] = (_Float16)(w2 * w1);
        }

        // ---- B operands: fp16 I-power tables (L2-hot, 16 halves each) ----
        int kg   = sc + klocal;
        int boff = bcol * N_NODES + kg + bBase;          // multiple of 16
        v16h B1 = *(const v16h*)(IP1 + boff);
        v16h B2 = *(const v16h*)(IP2 + boff);
        v16h B3 = *(const v16h*)(IP3 + boff);

        acc1 = __builtin_amdgcn_wmma_f32_16x16x32_f16(false, A1, false, B1,
                                                      (short)0, acc1, false, false);
        acc2 = __builtin_amdgcn_wmma_f32_16x16x32_f16(false, A2, false, B2,
                                                      (short)0, acc2, false, false);
        acc3 = __builtin_amdgcn_wmma_f32_16x16x32_f16(false, A3, false, B3,
                                                      (short)0, acc3, false, false);

        __syncthreads();   // everyone done reading ldsW[cur] before restage
    }

    // ---- cross-wave reduction + ps1 epilogue ----
#pragma unroll
    for (int v = 0; v < 8; ++v) {
        red[0][wave][v][lane] = acc1[v];
        red[1][wave][v][lane] = acc2[v];
        red[2][wave][v][lane] = acc3[v];
    }
    __syncthreads();

    if (wave == 0) {
        float p01 = psMatrix[1];   // expand_ps only touches the diagonal
#pragma unroll
        for (int v = 0; v < 8; ++v) {
            float y1 = 0.f, y2 = 0.f, y3 = 0.f;
#pragma unroll
            for (int w = 0; w < WAVES; ++w) {
                y1 += red[0][w][v][lane];
                y2 += red[1][w][v][lane];
                y3 += red[2][w][v][lane];
            }
            if (bcol < BATCH) {
                // C/D layout: VGPR v, lanes 0-15 -> M=v, lanes 16-31 -> M=v+8
                int j   = j0 + v + hiHalf * 8;
                float S = state[(size_t)bcol * 4 * N_NODES + j];
                float c = p01 * S;
                // sum_k log(1-x) ~= -(c*Y1 + c^2*Y2/2 + c^3*Y3/3), |x| <= 0.05
                float lp = -(c * y1 + 0.5f * c * c * y2
                             + (1.0f / 3.0f) * c * c * c * y3);
                ps1out[bcol * N_NODES + j] = 1.0f - expf(lp);
            }
        }
    }
}

// Elementwise: build psM, apply masked transpose-matmul + ps10 mix, sample.
__global__ void epi_finalize(const float* __restrict__ state,
                             const float* __restrict__ psMatrix,
                             const float* __restrict__ U,
                             const float* __restrict__ ps1,
                             float* __restrict__ out) {
    int idx = blockIdx.x * blockDim.x + threadIdx.x;      // over 8*4096
    if (idx >= BATCH * N_NODES) return;
    int b = idx >> 12;
    int n = idx & (N_NODES - 1);

    // psM = eye*(1 - rowsum(psMatrix)) + psMatrix
    float ps[4][4];
#pragma unroll
    for (int i = 0; i < 4; ++i) {
        float rs = 0.f;
#pragma unroll
        for (int j = 0; j < 4; ++j) { ps[i][j] = psMatrix[i * 4 + j]; rs += ps[i][j]; }
        ps[i][i] += 1.0f - rs;
    }

    const float* st = state + (size_t)b * 4 * N_NODES + n;
    float s0 = st[0];
    float s1 = st[1 * N_NODES];
    float s2 = st[2 * N_NODES];
    float s3 = st[3 * N_NODES];
    float p1 = ps1[b * N_NODES + n];

    // out[b,i,n] = S*ps10[i] + sum_{j>=1} psM[j][i]*state[b,j,n]  (row 0 masked)
    float P[4];
#pragma unroll
    for (int i = 0; i < 4; ++i) {
        float v = ps[1][i] * s1 + ps[2][i] * s2 + ps[3][i] * s3;
        if (i == 0) v += s0 * (1.0f - p1);
        if (i == 1) v += s0 * p1;
        P[i] = v;
    }

    // cumulative-threshold sampling
    float u = U[n];
    float* o = out + (size_t)b * 4 * N_NODES + n;
#pragma unroll
    for (int i = 0; i < 4; ++i) {
        u -= P[i];
        float s = (u < 0.0f) ? 1.0f : 0.0f;
        o[i * N_NODES] = s;
        u += s;
    }
}

extern "C" void kernel_launch(void* const* d_in, const int* in_sizes, int n_in,
                              void* d_out, int out_size, void* d_ws, size_t ws_size,
                              hipStream_t stream) {
    const float* state = (const float*)d_in[0];   // [8,4,4096]
    const float* W     = (const float*)d_in[1];   // [4096,4096]
    const float* psM   = (const float*)d_in[2];   // [4,4]
    const float* U     = (const float*)d_in[3];   // [4096]
    float* out = (float*)d_out;                   // [8,4,4096]

    char* ws = (char*)d_ws;
    float*    ps1 = (float*)ws;
    _Float16* IP1 = (_Float16*)(ws + 131072);
    _Float16* IP2 = (_Float16*)(ws + 262144);
    _Float16* IP3 = (_Float16*)(ws + 393216);

    hipLaunchKernelGGL(epi_prep_powers, dim3((16 * N_NODES) / 256), dim3(256), 0, stream,
                       state, IP1, IP2, IP3);
    hipLaunchKernelGGL(epi_infection_wmma, dim3(N_NODES / TILE_M), dim3(WAVES * 32), 0, stream,
                       W, state, psM, IP1, IP2, IP3, ps1);
    hipLaunchKernelGGL(epi_finalize, dim3((BATCH * N_NODES) / 256), dim3(256), 0, stream,
                       state, psM, U, ps1, out);
}
